// RandomForestNet_55113020342401
// MI455X (gfx1250) — matrix operand order repaired and checked by reading, not compiled, not run on hardware
//
#include <hip/hip_runtime.h>
#include <hip/hip_bf16.h>

#define B_ROWS 16384
#define F_DIM  256
#define E_DIM  512
#define DEPTHM1 7

typedef __attribute__((ext_vector_type(16))) _Float16 v16h;
typedef __attribute__((ext_vector_type(8)))  _Float16 v8h;
typedef __attribute__((ext_vector_type(4)))  _Float16 v4h;
typedef __attribute__((ext_vector_type(8)))  float    v8f;

union AB16 { v16h v; v8h h[2]; };

// ---------------------------------------------------------------------------
// Kernel 0: convert W0 [E,F] f32 -> f16 into workspace (0.5 MB -> 0.25 MB).
// 512*256 = 131072 elems = 32768 float4s; 128 blocks x 256 threads.
// ---------------------------------------------------------------------------
__global__ __launch_bounds__(256) void cvt_w0_f16(const float* __restrict__ w0,
                                                  _Float16* __restrict__ w0h) {
    int i = blockIdx.x * 256 + threadIdx.x;           // [0, 32768)
    float4 f = ((const float4*)w0)[i];
    v4h o = { (_Float16)f.x, (_Float16)f.y, (_Float16)f.z, (_Float16)f.w };
    ((v4h*)w0h)[i] = o;
}

// ---------------------------------------------------------------------------
// Kernel 1: fused GEMM (WMMA f16->f32) + bias/relu + 7-step chain + final
// linear + mean over E.  One workgroup = 32 rows of x, all 512 columns.
// 8 waves; wave w owns e in [w*64, w*64+64) as 4 n-tiles x 2 m-tiles.
// ---------------------------------------------------------------------------
#define LDS_STRIDE 264   // 256 + 8 halfword pad: bank step 4/row -> no conflicts

__device__ __forceinline__ v8f wmma_f16(v16h a, v16h b, v8f c) {
    return __builtin_amdgcn_wmma_f32_16x16x32_f16(
        /*neg_a=*/false, a, /*neg_b=*/false, b,
        /*c_mod=*/(short)0, c, /*reuse_a=*/false, /*reuse_b=*/false);
}

__global__ __launch_bounds__(256) void forest_fused(
    const float*    __restrict__ x,    // [B,F] f32
    const _Float16* __restrict__ w0h,  // [E,F] f16 (pre-converted)
    const float*    __restrict__ b0,   // [E]
    const float*    __restrict__ Wc,   // [7,E]
    const float*    __restrict__ bc,   // [7,E]
    const float*    __restrict__ Wf,   // [E]
    const float*    __restrict__ bf,   // [E]
    float*          __restrict__ out)  // [B]
{
    __shared__ _Float16 lds_x[32 * LDS_STRIDE];
    __shared__ float    lds_part[8][32];

    const int tid   = threadIdx.x;
    const int wave  = tid >> 5;
    const int lane  = tid & 31;
    const int half  = lane >> 4;        // 0: lanes 0-15, 1: lanes 16-31
    const int n16   = lane & 15;
    const int bbase = blockIdx.x * 32;  // 512 blocks * 32 rows = 16384
    const int ebase = wave * 64;        // this wave's 64 e-columns

    // ---- Stage x tile (32 x 256 f32) into LDS as f16, padded rows ----------
    // 2048 float4 loads; each thread does 8.
    #pragma unroll
    for (int i = 0; i < 8; ++i) {
        int idx = tid + i * 256;            // [0, 2048)
        int row = idx >> 6;                 // 64 float4 per row
        int k   = (idx & 63) << 2;
        float4 f = *(const float4*)(x + (size_t)(bbase + row) * F_DIM + k);
        v4h o = { (_Float16)f.x, (_Float16)f.y, (_Float16)f.z, (_Float16)f.w };
        *(v4h*)(&lds_x[row * LDS_STRIDE + k]) = o;
    }
    __syncthreads();

    // ---- GEMM: acc[et][bt] = x_tile(bt) @ W0^T(et), K = 256 ----------------
    v8f acc[4][2];
    #pragma unroll
    for (int et = 0; et < 4; ++et)
        #pragma unroll
        for (int bt = 0; bt < 2; ++bt)
            acc[et][bt] = (v8f)(0.0f);

    // Per-lane B row pointers (K chunks {half*8, half*8+16} per k-step).
    const _Float16* brow[4];
    #pragma unroll
    for (int et = 0; et < 4; ++et)
        brow[et] = w0h + (size_t)(ebase + et * 16 + n16) * F_DIM + half * 8;

    for (int k0 = 0; k0 < F_DIM; k0 += 32) {
        // A fragments: lane (half,n16) -> row n16, K chunks {half*8, half*8+16}
        AB16 a0, a1;
        {
            const _Float16* p0 = &lds_x[(0 * 16 + n16) * LDS_STRIDE + k0 + half * 8];
            const _Float16* p1 = &lds_x[(1 * 16 + n16) * LDS_STRIDE + k0 + half * 8];
            a0.h[0] = *(const v8h*)(p0);
            a0.h[1] = *(const v8h*)(p0 + 16);
            a1.h[0] = *(const v8h*)(p1);
            a1.h[1] = *(const v8h*)(p1 + 16);
        }
        #pragma unroll
        for (int et = 0; et < 4; ++et) {
            AB16 bfrag;
            const _Float16* q = brow[et] + k0;
            bfrag.h[0] = *(const v8h*)(q);
            bfrag.h[1] = *(const v8h*)(q + 16);
            acc[et][0] = wmma_f16(a0.v, bfrag.v, acc[et][0]);
            acc[et][1] = wmma_f16(a1.v, bfrag.v, acc[et][1]);
        }
    }

    // ---- Fused epilogue: bias + relu + 7x (relu(h*w+b)) + h*Wf+bf, row sum -
    float rs0[8], rs1[8];
    #pragma unroll
    for (int r = 0; r < 8; ++r) { rs0[r] = 0.0f; rs1[r] = 0.0f; }

    #pragma unroll
    for (int et = 0; et < 4; ++et) {
        const int e = ebase + et * 16 + n16;   // this lane's column
        const float b0e = b0[e];
        const float wfe = Wf[e];
        const float bfe = bf[e];
        float wcd[DEPTHM1], bcd[DEPTHM1];
        #pragma unroll
        for (int d = 0; d < DEPTHM1; ++d) {
            wcd[d] = Wc[d * E_DIM + e];
            bcd[d] = bc[d * E_DIM + e];
        }
        #pragma unroll
        for (int bt = 0; bt < 2; ++bt) {
            float* rs = bt ? rs1 : rs0;
            #pragma unroll
            for (int r = 0; r < 8; ++r) {
                float h = acc[et][bt][r] + b0e;
                h = fmaxf(h, 0.0f);
                #pragma unroll
                for (int d = 0; d < DEPTHM1; ++d)
                    h = fmaxf(fmaf(h, wcd[d], bcd[d]), 0.0f);
                rs[r] = fmaf(h, wfe, bfe + rs[r]);
            }
        }
    }

    // ---- Wave reduction: butterfly within each 16-lane half (wave32) -------
    #pragma unroll
    for (int m = 1; m <= 8; m <<= 1) {
        #pragma unroll
        for (int r = 0; r < 8; ++r) {
            rs0[r] += __shfl_xor(rs0[r], m, 32);
            rs1[r] += __shfl_xor(rs1[r], m, 32);
        }
    }
    // lanes 0 / 16 now hold sums for rows {half*8 .. half*8+7} of each m-tile
    if (n16 == 0) {
        #pragma unroll
        for (int r = 0; r < 8; ++r) {
            lds_part[wave][0 * 16 + half * 8 + r] = rs0[r];
            lds_part[wave][1 * 16 + half * 8 + r] = rs1[r];
        }
    }
    __syncthreads();

    // ---- Cross-wave reduce (8 waves x 64 cols each = all 512), mean --------
    if (tid < 32) {
        float s = 0.0f;
        #pragma unroll
        for (int w = 0; w < 8; ++w) s += lds_part[w][tid];
        out[bbase + tid] = s * (1.0f / (float)E_DIM);
    }
}

// ---------------------------------------------------------------------------
extern "C" void kernel_launch(void* const* d_in, const int* in_sizes, int n_in,
                              void* d_out, int out_size, void* d_ws, size_t ws_size,
                              hipStream_t stream) {
    const float* x  = (const float*)d_in[0];   // [16384,256]
    const float* W0 = (const float*)d_in[1];   // [512,256]
    const float* b0 = (const float*)d_in[2];   // [512]
    const float* Wc = (const float*)d_in[3];   // [7,512]
    const float* bc = (const float*)d_in[4];   // [7,512]
    const float* Wf = (const float*)d_in[5];   // [512]
    const float* bf = (const float*)d_in[6];   // [512]
    float* out = (float*)d_out;                // [16384]

    _Float16* w0h = (_Float16*)d_ws;           // 512*256*2 = 256 KB scratch

    cvt_w0_f16<<<128, 256, 0, stream>>>(W0, w0h);
    forest_fused<<<B_ROWS / 32, 256, 0, stream>>>(x, w0h, b0, Wc, bc, Wf, bf, out);
}
